// GraphMultisetTransformer_24687472017553
// MI455X (gfx1250) — matrix operand order repaired and checked by reading, not compiled, not run on hardware
//
#include <hip/hip_runtime.h>

// ---------------------------------------------------------------------------
// Graph Multiset Transformer for MI455X (gfx1250, wave32, WMMA)
//   N=20000 nodes, E=320000 edges, D=128, K=2000 seeds
// All dense GEMMs: v_wmma_f32_16x16x32_bf16 with BOTH operands stored bf16 in
// memory -> inner loop is 6x global_load_b128 + 2x v_wmma, zero conversions,
// zero bounds checks (K padded to a multiple of 32 with zeros where needed).
// ---------------------------------------------------------------------------

#define N_NODES 20000
#define N_EDGES 320000
#define DIM     128
#define K_SEEDS 2000
#define K_PAD   2016   // K_SEEDS rounded up to a multiple of 32 (zero padded)

typedef __attribute__((ext_vector_type(16))) __bf16         v16bf;
typedef __attribute__((ext_vector_type(8)))  float          v8f;
typedef __attribute__((ext_vector_type(8)))  unsigned short us8v;
typedef __attribute__((ext_vector_type(16))) unsigned short us16v;

__device__ __forceinline__ __bf16 f2bf(float f) {
  unsigned u = __float_as_uint(f);
  unsigned r = (u + 0x7FFFu + ((u >> 16) & 1u)) >> 16;   // round-to-nearest-even
  unsigned short s = (unsigned short)r;
  __bf16 b;
  __builtin_memcpy(&b, &s, sizeof(b));
  return b;
}

// ---------------------------------------------------------------------------
// WMMA GEMM: C[M x Ncols] = act( scale * A[M x K] * B[K x Ncols] + bias ) (+res)
//   A: bf16 row-major M x K.   B: bf16 TRANSPOSED row-major Ncols x K.
// One wave -> 16x32 tile (2 WMMAs / K-step). M%16==0, Ncols%32==0, Kdim%32==0.
// ---------------------------------------------------------------------------
__global__ void wmma_gemm_kernel(const __bf16* __restrict__ A, int lda,
                                 const __bf16* __restrict__ Bt, int ldb,
                                 const float* __restrict__ bias,
                                 const float* __restrict__ residual,
                                 float* __restrict__ C,
                                 __bf16* __restrict__ Cbf, int ldc,
                                 int M, int Ncols, int Kdim,
                                 float scale, int do_relu)
{
  const int lane = threadIdx.x & 31;
  const int wave = threadIdx.x >> 5;
  const int ntN  = Ncols >> 5;
  const long tile   = (long)blockIdx.x * (blockDim.x >> 5) + wave;
  const long ntiles = (long)(M >> 4) * ntN;
  if (tile >= ntiles) return;                 // wave-uniform: EXEC stays full
  const int tm = (int)(tile / ntN) * 16;
  const int tn = (int)(tile % ntN) * 32;

  const int h   = lane >> 4;                  // lane half (0/1)
  const int l15 = lane & 15;

  const __bf16* Arow = A + (size_t)(tm + l15) * lda;
  const __bf16* B0   = Bt + (size_t)(tn + l15) * ldb + h * 16;  // cols tn..+15
  const __bf16* B1   = B0 + (size_t)16 * ldb;                   // cols tn+16..+31

  v8f acc0 = {}, acc1 = {};

  for (int kb = 0; kb < Kdim; kb += 32) {
    // A lane data: one run of 16 (lane half selects k 0..15 / 16..31)
    us8v a0  = *(const us8v*)(Arow + kb + h * 16);
    us8v a1  = *(const us8v*)(Arow + kb + h * 16 + 8);
    // B lane data: one run of 16 per tile (lane half selects k 0..15 / 16..31)
    us8v b00 = *(const us8v*)(B0 + kb);
    us8v b01 = *(const us8v*)(B0 + kb + 8);
    us8v b10 = *(const us8v*)(B1 + kb);
    us8v b11 = *(const us8v*)(B1 + kb + 8);
    us16v au  = __builtin_shufflevector(a0,  a1,  0,1,2,3,4,5,6,7,8,9,10,11,12,13,14,15);
    us16v bu0 = __builtin_shufflevector(b00, b01, 0,1,2,3,4,5,6,7,8,9,10,11,12,13,14,15);
    us16v bu1 = __builtin_shufflevector(b10, b11, 0,1,2,3,4,5,6,7,8,9,10,11,12,13,14,15);
    v16bf a, b0, b1;
    __builtin_memcpy(&a,  &au,  32);
    __builtin_memcpy(&b0, &bu0, 32);
    __builtin_memcpy(&b1, &bu1, 32);
    acc0 = __builtin_amdgcn_wmma_f32_16x16x32_bf16(false, a, false, b0, (short)0, acc0, false, false);
    acc1 = __builtin_amdgcn_wmma_f32_16x16x32_bf16(false, a, false, b1, (short)0, acc1, false, false);
  }

#pragma unroll
  for (int r = 0; r < 8; ++r) {               // C/D rows: r + 8*laneHalf
    const int rr = tm + r + h * 8;
    const int c0 = tn + l15;
    const int c1 = c0 + 16;
    float v0 = acc0[r] * scale, v1 = acc1[r] * scale;
    if (bias)     { v0 += bias[c0];  v1 += bias[c1]; }
    if (do_relu)  { v0 = fmaxf(v0, 0.f); v1 = fmaxf(v1, 0.f); }
    if (residual) { v0 += residual[(size_t)rr * ldc + c0];
                    v1 += residual[(size_t)rr * ldc + c1]; }
    C[(size_t)rr * ldc + c0] = v0;
    C[(size_t)rr * ldc + c1] = v1;
    if (Cbf) { Cbf[(size_t)rr * ldc + c0] = f2bf(v0);
               Cbf[(size_t)rr * ldc + c1] = f2bf(v1); }
  }
}

// ---------------------------------------------------------------------------
// tiled transpose f32 -> bf16 (coalesced both ways via 32x33 LDS tile)
// src R x C (f32)  ->  dst C x ldo (bf16), columns [R, ldo) zero-filled.
// grid.y must cover ldo (ceil(ldo/32)).
// ---------------------------------------------------------------------------
__global__ void k_transpose_bf16(const float* __restrict__ src, __bf16* __restrict__ dst,
                                 int R, int C, int ldo) {
  __shared__ float t[32][33];
  const int c0 = blockIdx.x * 32, r0 = blockIdx.y * 32;
  const int tx = threadIdx.x, ty = threadIdx.y;
#pragma unroll
  for (int i = 0; i < 32; i += 8) {
    int r = r0 + ty + i, c = c0 + tx;
    t[ty + i][tx] = (r < R && c < C) ? src[(size_t)r * C + c] : 0.f;
  }
  __syncthreads();
#pragma unroll
  for (int i = 0; i < 32; i += 8) {
    int rr = c0 + ty + i, cc = r0 + tx;
    if (rr < C && cc < ldo) dst[(size_t)rr * ldo + cc] = f2bf(t[tx][ty + i]);
  }
}

// elementwise f32 -> bf16
__global__ void k_cvt_bf16(const float* __restrict__ src, __bf16* __restrict__ dst, long n) {
  long i = (long)blockIdx.x * blockDim.x + threadIdx.x;
  if (i < n) dst[i] = f2bf(src[i]);
}

// ---------------------------------------------------------------------------
// GCN support kernels
// ---------------------------------------------------------------------------
__global__ void k_zero(float* __restrict__ p, long n) {
  long i = (long)blockIdx.x * blockDim.x + threadIdx.x;
  if (i < n) p[i] = 0.f;
}
__global__ void k_deg(const int* __restrict__ dst, float* __restrict__ deg, int e) {
  int i = blockIdx.x * blockDim.x + threadIdx.x;
  if (i < e) atomicAdd(&deg[dst[i]], 1.f);
}
__global__ void k_dinv(const float* __restrict__ deg, float* __restrict__ dinv, int n) {
  int i = blockIdx.x * blockDim.x + threadIdx.x;
  if (i < n) dinv[i] = rsqrtf(fmaxf(deg[i] + 1.f, 1e-12f));
}
__global__ void k_selfloop(const float* __restrict__ H, const float* __restrict__ dinv,
                           float* __restrict__ out, long cnt) {
  long i = (long)blockIdx.x * blockDim.x + threadIdx.x;
  if (i < cnt) {
    int n = (int)(i >> 7);
    float di = dinv[n];
    out[i] = di * di * H[i];
  }
}
__global__ void k_scatter(const float* __restrict__ H, const float* __restrict__ dinv,
                          const int* __restrict__ src, const int* __restrict__ dst,
                          float* __restrict__ out) {
  const int e = blockIdx.x;
  const int f = threadIdx.x;
  const int s = src[e], d = dst[e];
  const float nrm = dinv[s] * dinv[d];
  atomicAdd(&out[(size_t)d * DIM + f], nrm * H[(size_t)s * DIM + f]);
}
// bias (+relu); writes fp32 in place and optional bf16 copy
__global__ void k_bias_act(float* __restrict__ X, __bf16* __restrict__ Xbf,
                           const float* __restrict__ b, long cnt, int relu) {
  long i = (long)blockIdx.x * blockDim.x + threadIdx.x;
  if (i < cnt) {
    float v = X[i] + b[i & (DIM - 1)];
    if (relu) v = fmaxf(v, 0.f);
    X[i] = v;
    if (Xbf) Xbf[i] = f2bf(v);
  }
}

// online softmax over seed (row) dim of A[K x N]; thread per column (coalesced)
// writes fp32 in place + bf16 copy
__global__ void k_softmax_col(float* __restrict__ A, __bf16* __restrict__ Abf,
                              int rows, int cols) {
  int n = blockIdx.x * blockDim.x + threadIdx.x;
  if (n >= cols) return;
  float m = -3.4e38f, s = 0.f;
  for (int k = 0; k < rows; ++k) {
    float v = A[(size_t)k * cols + n];
    float nm = fmaxf(m, v);
    s = s * __expf(m - nm) + __expf(v - nm);
    m = nm;
  }
  float inv = 1.f / s;
  for (int k = 0; k < rows; ++k) {
    size_t idx = (size_t)k * cols + n;
    float r = __expf(A[idx] - m) * inv;
    A[idx] = r;
    Abf[idx] = f2bf(r);
  }
}

// LayerNorm over last dim (=128); one 128-thread block per row; bf16 twin out
__global__ void k_layernorm(float* __restrict__ X, __bf16* __restrict__ Xbf,
                            const float* __restrict__ g, const float* __restrict__ b) {
  __shared__ float red[DIM];
  const int row = blockIdx.x;
  const int f = threadIdx.x;
  float v = X[(size_t)row * DIM + f];
  red[f] = v; __syncthreads();
  for (int s = DIM / 2; s > 0; s >>= 1) { if (f < s) red[f] += red[f + s]; __syncthreads(); }
  const float mean = red[0] * (1.f / DIM);
  __syncthreads();
  const float dv = v - mean;
  red[f] = dv * dv; __syncthreads();
  for (int s = DIM / 2; s > 0; s >>= 1) { if (f < s) red[f] += red[f + s]; __syncthreads(); }
  const float var = red[0] * (1.f / DIM);
  float o = dv * rsqrtf(var + 1e-5f) * g[f] + b[f];
  X[(size_t)row * DIM + f] = o;
  if (Xbf) Xbf[(size_t)row * DIM + f] = f2bf(o);
}

// ---------------------------------------------------------------------------
// host-side helpers
// ---------------------------------------------------------------------------
static void gemm(const __bf16* A, int lda, const __bf16* Bt, int ldb,
                 const float* bias, const float* residual,
                 float* C, __bf16* Cbf, int ldc,
                 int M, int Nc, int Kd, float scale, int relu, hipStream_t s)
{
  long tiles = (long)(M / 16) * (Nc / 32);
  int blocks = (int)((tiles + 3) / 4);               // 4 waves / block
  wmma_gemm_kernel<<<blocks, 128, 0, s>>>(A, lda, Bt, ldb, bias, residual,
                                          C, Cbf, ldc, M, Nc, Kd, scale, relu);
}

// out = GCN(Xbf, W^T, b) (optional relu); fp32 out + optional bf16 twin
static void gcn(const __bf16* Xbf, const __bf16* WTb, const float* b,
                const int* src, const int* dst, const float* dinv,
                float* tmp, float* outf, __bf16* outbf, int relu, hipStream_t s)
{
  gemm(Xbf, DIM, WTb, DIM, nullptr, nullptr, tmp, nullptr, DIM,
       N_NODES, DIM, DIM, 1.f, 0, s);
  const long cnt = (long)N_NODES * DIM;
  k_selfloop<<<(int)((cnt + 255) / 256), 256, 0, s>>>(tmp, dinv, outf, cnt);
  k_scatter<<<N_EDGES, DIM, 0, s>>>(tmp, dinv, src, dst, outf);
  k_bias_act<<<(int)((cnt + 255) / 256), 256, 0, s>>>(outf, outbf, b, cnt, relu);
}

extern "C" void kernel_launch(void* const* d_in, const int* in_sizes, int n_in,
                              void* d_out, int out_size, void* d_ws, size_t ws_size,
                              hipStream_t stream)
{
  const float* x  = (const float*)d_in[0];
  const int*   ei = (const int*)d_in[1];
  const int* src = ei;
  const int* dst = ei + N_EDGES;
  const float* W1=(const float*)d_in[3];  const float* b1=(const float*)d_in[4];
  const float* W2=(const float*)d_in[5];  const float* b2=(const float*)d_in[6];
  const float* S =(const float*)d_in[7];
  const float* Wq=(const float*)d_in[8];  const float* bq=(const float*)d_in[9];
  const float* Wk=(const float*)d_in[10]; const float* bk=(const float*)d_in[11];
  const float* Wv=(const float*)d_in[12]; const float* bv=(const float*)d_in[13];
  const float* Wo=(const float*)d_in[14]; const float* bo=(const float*)d_in[15];
  const float* g0=(const float*)d_in[16]; const float* be0=(const float*)d_in[17];
  const float* g1=(const float*)d_in[18]; const float* be1=(const float*)d_in[19];
  const float* Wl=(const float*)d_in[20]; const float* bl=(const float*)d_in[21];
  const float* W3=(const float*)d_in[22]; const float* b3=(const float*)d_in[23];
  const float* W4=(const float*)d_in[24]; const float* b4=(const float*)d_in[25];
  const float* W5=(const float*)d_in[26]; const float* b5=(const float*)d_in[27];
  float* out = (float*)d_out;

  // ---- workspace layout (all carve sizes are multiples of 4 floats) -------
  float* ws = (float*)d_ws;
  size_t off = 0;
  auto alloc  = [&](size_t n) { float* p = ws + off; off += n; return p; };
  auto allocb = [&](size_t nbf) { return (__bf16*)alloc(nbf / 2); }; // nbf elems
  const size_t ND  = (size_t)N_NODES * DIM;
  const size_t KD  = (size_t)K_SEEDS * DIM;
  const size_t KN  = (size_t)K_SEEDS * N_NODES;
  const size_t NKP = (size_t)N_NODES * K_PAD;
  const size_t DKP = (size_t)DIM * K_PAD;

  float* deg  = alloc(N_NODES);
  float* dinv = alloc(N_NODES);
  float* tmp  = alloc(ND);          // pre-aggregation scratch (f32)
  float* buf0 = alloc(ND);          // f32 aggregation targets
  float* buf2 = alloc(ND);
  float* buf3 = alloc(ND);          // Vd f32 (transpose source)
  float* Q    = alloc(KD);
  float* outb = alloc(KD);          // out, later xl (f32)
  float* out2 = alloc(KD);
  float* Amat = alloc(KN);          // f32 scores -> softmax in place

  __bf16* WTb[10];
  for (int i = 0; i < 10; ++i) WTb[i] = allocb((size_t)DIM * DIM);
  __bf16* xbf    = allocb(ND);
  __bf16* h1bf   = allocb(ND);
  __bf16* h2bf   = allocb(ND);
  __bf16* Kdbf   = allocb(ND);
  __bf16* VdTbf  = allocb(ND);      // [D x N]
  __bf16* xoutbf = allocb(ND);
  __bf16* h3bf   = allocb(ND);
  __bf16* h4bf   = allocb(ND);
  __bf16* Sbf    = allocb(KD);
  __bf16* Qbf    = allocb(KD);
  __bf16* outbbf = allocb(KD);
  __bf16* out2bf = allocb(KD);
  __bf16* xlTbf  = allocb(DKP);     // [D x K_PAD], zero padded
  __bf16* Abf    = allocb(KN);      // softmax, bf16 row-major [K x N]
  __bf16* AbfT   = allocb(NKP);     // softmax^T bf16 [N x K_PAD], zero padded
  (void)in_sizes; (void)n_in; (void)out_size; (void)ws_size;

  dim3 tb(32, 8);

  // ---- weights: transpose straight to bf16 (tiny) -------------------------
  const float* Wsrc[10] = { W1, W2, Wq, Wk, Wv, Wo, Wl, W3, W4, W5 };
  for (int i = 0; i < 10; ++i)
    k_transpose_bf16<<<dim3(4, 4), tb, 0, stream>>>(Wsrc[i], WTb[i], DIM, DIM, DIM);

  // ---- degree / symmetric normalization -----------------------------------
  k_zero<<<(N_NODES + 255) / 256, 256, 0, stream>>>(deg, N_NODES);
  k_deg <<<(N_EDGES + 255) / 256, 256, 0, stream>>>(dst, deg, N_EDGES);
  k_dinv<<<(N_NODES + 255) / 256, 256, 0, stream>>>(deg, dinv, N_NODES);

  // ---- inputs to bf16 ------------------------------------------------------
  k_cvt_bf16<<<(int)((ND + 255) / 256), 256, 0, stream>>>(x, xbf, (long)ND);
  k_cvt_bf16<<<(int)((KD + 255) / 256), 256, 0, stream>>>(S, Sbf, (long)KD);

  // ---- two GCN encoder layers ---------------------------------------------
  gcn(xbf,  WTb[0], b1, src, dst, dinv, tmp, buf0, h1bf, 1, stream);
  gcn(h1bf, WTb[1], b2, src, dst, dinv, tmp, buf2, h2bf, 1, stream);

  // ---- PMA attention -------------------------------------------------------
  gemm(Sbf, DIM, WTb[2], DIM, bq, nullptr, Q, Qbf, DIM,
       K_SEEDS, DIM, DIM, 1.f, 0, stream);                     // Q = S@Wq+bq
  gcn(h2bf, WTb[3], bk, src, dst, dinv, tmp, buf0, Kdbf, 0, stream);    // Kd
  gcn(h2bf, WTb[4], bv, src, dst, dinv, tmp, buf3, nullptr, 0, stream); // Vd f32

  // score[K x N] = (Q @ Kd^T)/sqrt(D); Kd rows ARE the transposed-B layout
  gemm(Qbf, DIM, Kdbf, DIM, nullptr, nullptr, Amat, nullptr, N_NODES,
       K_SEEDS, N_NODES, DIM, 0.08838834764831845f, 0, stream);
  k_softmax_col<<<(N_NODES + 255) / 256, 256, 0, stream>>>(Amat, Abf, K_SEEDS, N_NODES);

  // out = Q + A @ Vd  (B side: Vd^T bf16 [D x N])
  k_transpose_bf16<<<dim3((DIM + 31) / 32, (N_NODES + 31) / 32), tb, 0, stream>>>(
      buf3, VdTbf, N_NODES, DIM, N_NODES);
  gemm(Abf, N_NODES, VdTbf, N_NODES, nullptr, Q, outb, nullptr, DIM,
       K_SEEDS, DIM, N_NODES, 1.f, 0, stream);
  k_layernorm<<<K_SEEDS, DIM, 0, stream>>>(outb, outbbf, g0, be0);

  // out2 = out + relu(out@Wo + bo)
  gemm(outbbf, DIM, WTb[5], DIM, bo, outb, out2, nullptr, DIM,
       K_SEEDS, DIM, DIM, 1.f, 1, stream);
  k_layernorm<<<K_SEEDS, DIM, 0, stream>>>(out2, out2bf, g1, be1);

  // xl = out2@Wl + bl  (f32 into outb, then transpose to bf16 [D x K_PAD])
  gemm(out2bf, DIM, WTb[6], DIM, bl, nullptr, outb, nullptr, DIM,
       K_SEEDS, DIM, DIM, 1.f, 0, stream);
  k_transpose_bf16<<<dim3((DIM + 31) / 32, (K_PAD + 31) / 32), tb, 0, stream>>>(
      outb, xlTbf, K_SEEDS, DIM, K_PAD);

  // x_out[N x D] = A^T @ xl : A^T bf16 [N x K_PAD], xl^T bf16 [D x K_PAD]
  k_transpose_bf16<<<dim3((N_NODES + 31) / 32, (K_PAD + 31) / 32), tb, 0, stream>>>(
      Amat, AbfT, K_SEEDS, N_NODES, K_PAD);
  gemm(AbfT, K_PAD, xlTbf, K_PAD, nullptr, nullptr, buf2, xoutbf, DIM,
       N_NODES, DIM, K_PAD, 1.f, 0, stream);

  // ---- decoder GCNs --------------------------------------------------------
  gcn(xoutbf, WTb[7], b3, src, dst, dinv, tmp, buf0, h3bf, 1, stream);
  gcn(h3bf,   WTb[8], b4, src, dst, dinv, tmp, buf3, h4bf, 1, stream);
  gcn(h4bf,   WTb[9], b5, src, dst, dinv, tmp, out,  nullptr, 0, stream);
}
